// CausalSelfAttention_50525995270469
// MI455X (gfx1250) — matrix operand order repaired and checked by reading, not compiled, not run on hardware
//
#include <hip/hip_runtime.h>
#include <hip/hip_bf16.h>

// ---------------- problem constants ----------------
#define BB    8
#define TSEQ  1024
#define CEMB  768
#define NH    12
#define HD    64
#define C3    (3 * CEMB)          // 2304
#define MROWS (BB * TSEQ)         // 8192

// ---------------- WMMA fragment types ----------------
typedef __bf16 v16bf __attribute__((ext_vector_type(16)));
typedef float  v8f   __attribute__((ext_vector_type(8)));

union FragU { v16bf bf; unsigned int u[8]; };

// bf16 round-to-nearest-even from f32 (bit pattern)
__device__ __forceinline__ unsigned short f2bf(float f) {
  unsigned int u = __float_as_uint(f);
  u = (u + 0x7FFFu + ((u >> 16) & 1u)) >> 16;
  return (unsigned short)u;
}

// ---- gfx1250 async global->LDS copy (16B per lane), tracked by ASYNCcnt ----
// LDS_ADDR aperture rule: low 32 bits of the flat shared pointer are the LDS
// byte offset, which is what the VDST operand of the async op wants.
__device__ __forceinline__ void async_ld_b128(void* lds_p, const void* g_p) {
  unsigned int loff = (unsigned int)(size_t)lds_p;
  asm volatile("global_load_async_to_lds_b128 %0, %1, off"
               :: "v"(loff), "v"(g_p) : "memory");
}
__device__ __forceinline__ void wait_async0() {
  asm volatile("s_wait_asynccnt 0x0" ::: "memory");
}

// A fragment: 16x32 bf16 (MxK) from row-major LDS tile, lda in u16 elems (even).
__device__ __forceinline__ v16bf load_frag_a(const unsigned short* s, int lda) {
  const int lane = threadIdx.x & 31;
  const int row  = lane & 15;
  const int hi4  = (lane >> 4) << 2;   // 0 or 4 (dword offset)
  const unsigned int* p = (const unsigned int*)(s + row * lda);
  FragU f;
#pragma unroll
  for (int j = 0; j < 4; ++j) {
    f.u[j]     = p[hi4 + j];           // K = 2j,2j+1 (+8 for hi half)
    f.u[4 + j] = p[8 + hi4 + j];       // K = 16+2j,.. (+8 for hi half)
  }
  return f.bf;
}

// B fragment: 32x16 bf16 (KxN) from row-major LDS tile B[k][n], ldb in u16 elems.
__device__ __forceinline__ v16bf load_frag_b(const unsigned short* s, int ldb) {
  const int lane = threadIdx.x & 31;
  const int n = lane & 15;
  const int hk = (lane >> 4) << 3;
  FragU f;
#pragma unroll
  for (int j = 0; j < 4; ++j) {
    int k0 = hk + 2 * j, k1 = 16 + hk + 2 * j;
    f.u[j]     = (unsigned int)s[k0 * ldb + n] | ((unsigned int)s[(k0 + 1) * ldb + n] << 16);
    f.u[4 + j] = (unsigned int)s[k1 * ldb + n] | ((unsigned int)s[(k1 + 1) * ldb + n] << 16);
  }
  return f.bf;
}

__device__ __forceinline__ v8f wmma_bf16(v16bf a, v16bf b, v8f c) {
  return __builtin_amdgcn_wmma_f32_16x16x32_bf16(false, a, false, b, (short)0, c,
                                                 false, false);
}

// ---------------- fp32 -> bf16 cast ----------------
__global__ void __launch_bounds__(256) cvt_bf16(const float* __restrict__ src,
                                                unsigned short* __restrict__ dst) {
  const int i = blockIdx.x * 256 + threadIdx.x;
  dst[i] = f2bf(src[i]);
}

// ---------------- QKV GEMM: [8192,768] x [768,2304] -> Q/K/V [B,H,T,D] bf16 ----
// Block tile 128x64, 4 waves, each wave: 32-row strip x 64 cols = 8 WMMA/K-chunk.
__global__ void __launch_bounds__(128)
gemm_qkv(const unsigned short* __restrict__ X, const unsigned short* __restrict__ W,
         unsigned short* __restrict__ Qo, unsigned short* __restrict__ Ko,
         unsigned short* __restrict__ Vo) {
  __shared__ __attribute__((aligned(16))) unsigned short As[128 * 32];
  __shared__ __attribute__((aligned(16))) unsigned short Bs[32 * 64];
  const int tid = threadIdx.x, lane = tid & 31, wave = tid >> 5;
  const int m0 = blockIdx.y * 128, n0 = blockIdx.x * 64;

  const v8f zf = {0.f, 0.f, 0.f, 0.f, 0.f, 0.f, 0.f, 0.f};
  v8f acc[2][4];
#pragma unroll
  for (int mt = 0; mt < 2; ++mt)
#pragma unroll
    for (int nt = 0; nt < 4; ++nt) acc[mt][nt] = zf;

  for (int kt = 0; kt < CEMB / 32; ++kt) {
    // A tile 128x32 u16 = 512 x 16B chunks (4/row), async DMA to LDS
#pragma unroll
    for (int i = 0; i < 4; ++i) {
      int q = tid + i * 128;
      int r = q >> 2, c = q & 3;
      async_ld_b128(As + r * 32 + c * 8, X + (size_t)(m0 + r) * CEMB + kt * 32 + c * 8);
    }
    // B tile 32x64 u16 = 256 x 16B chunks (8/row)
#pragma unroll
    for (int i = 0; i < 2; ++i) {
      int q = tid + i * 128;
      int r = q >> 3, c = q & 7;
      async_ld_b128(Bs + r * 64 + c * 8, W + (size_t)(kt * 32 + r) * C3 + n0 + c * 8);
    }
    if (kt + 1 < CEMB / 32)
      __builtin_prefetch((const void*)(W + (size_t)((kt + 1) * 32) * C3 + n0 + lane * 8), 0, 1);
    wait_async0();
    __syncthreads();

    v16bf a0 = load_frag_a(As + (wave * 32 +  0) * 32, 32);
    v16bf a1 = load_frag_a(As + (wave * 32 + 16) * 32, 32);
#pragma unroll
    for (int nt = 0; nt < 4; ++nt) {
      v16bf b = load_frag_b(Bs + nt * 16, 64);
      acc[0][nt] = wmma_bf16(a0, b, acc[0][nt]);
      acc[1][nt] = wmma_bf16(a1, b, acc[1][nt]);
    }
    __syncthreads();
  }

  const int hi = lane >> 4, nl = lane & 15;
#pragma unroll
  for (int mt = 0; mt < 2; ++mt)
#pragma unroll
    for (int nt = 0; nt < 4; ++nt)
#pragma unroll
      for (int r = 0; r < 8; ++r) {
        int m = m0 + wave * 32 + mt * 16 + r + 8 * hi;
        int n = n0 + nt * 16 + nl;
        int which = n / CEMB, c = n % CEMB;
        int h = c >> 6, d = c & 63;
        int bidx = m >> 10, t = m & 1023;
        unsigned short* dst = (which == 0) ? Qo : ((which == 1) ? Ko : Vo);
        dst[(((size_t)bidx * NH + h) * TSEQ + t) * HD + d] = f2bf(acc[mt][nt][r]);
      }
}

// ---------------- Flash attention: one block per (b,h, 64 query rows) --------
__global__ void __launch_bounds__(128)
attn_flash(const unsigned short* __restrict__ Q, const unsigned short* __restrict__ K,
           const unsigned short* __restrict__ V, unsigned short* __restrict__ Y) {
  __shared__ __attribute__((aligned(16))) unsigned short Qs[64 * 64];
  __shared__ __attribute__((aligned(16))) unsigned short Kt[64 * 64];  // transposed [d][t]
  __shared__ __attribute__((aligned(16))) unsigned short Vs[64 * 64];  // [t][d]
  __shared__ __attribute__((aligned(16))) unsigned short Ps[4 * 16 * 64];

  const int tid = threadIdx.x, lane = tid & 31, wave = tid >> 5;
  const int hi = lane >> 4, nl = lane & 15;
  const int bh = blockIdx.y;            // b*NH + h
  const int b = bh / NH, h = bh % NH;
  const int qb = blockIdx.x, q0 = qb * 64;

  const unsigned short* Qg = Q + (size_t)bh * TSEQ * HD;
  const unsigned short* Kg = K + (size_t)bh * TSEQ * HD;
  const unsigned short* Vg = V + (size_t)bh * TSEQ * HD;
  const unsigned int* Kd = (const unsigned int*)Kg;

  // stage Q tile 64x64 u16 = 512 x 16B chunks (8/row) via async DMA
#pragma unroll
  for (int i = 0; i < 4; ++i) {
    int q = tid + i * 128;
    int r = q >> 3, c = q & 7;
    async_ld_b128(Qs + r * 64 + c * 8, Qg + (size_t)(q0 + r) * HD + c * 8);
  }
  wait_async0();
  __syncthreads();

  const v16bf aq0 = load_frag_a(Qs + (wave * 16) * 64 + 0, 64);
  const v16bf aq1 = load_frag_a(Qs + (wave * 16) * 64 + 32, 64);

  const v8f zf = {0.f, 0.f, 0.f, 0.f, 0.f, 0.f, 0.f, 0.f};
  v8f o[4];
  float mrow[8], lrow[8];
#pragma unroll
  for (int i = 0; i < 4; ++i) o[i] = zf;
#pragma unroll
  for (int r = 0; r < 8; ++r) { mrow[r] = -1e30f; lrow[r] = 0.f; }
  const float scale = 0.125f;           // 1/sqrt(64)

  for (int kb = 0; kb <= qb; ++kb) {
    // V tile: straight async DMA; K tile: manual transposed store [d][t]
#pragma unroll
    for (int i = 0; i < 4; ++i) {
      int q = tid + i * 128;
      int r = q >> 3, c = q & 7;
      async_ld_b128(Vs + r * 64 + c * 8, Vg + (size_t)(kb * 64 + r) * HD + c * 8);
    }
#pragma unroll
    for (int i = 0; i < 16; ++i) {
      int di = tid + i * 128;
      int r = di >> 5, c = di & 31;
      unsigned int kw = Kd[(kb * 64 + r) * 32 + c];
      Kt[(2 * c + 0) * 64 + r] = (unsigned short)(kw & 0xFFFFu);
      Kt[(2 * c + 1) * 64 + r] = (unsigned short)(kw >> 16);
    }
    wait_async0();
    __syncthreads();

    // scores S = scale * Q Kt  (16 rows x 64 key-cols per wave)
    v8f s[4];
#pragma unroll
    for (int nt = 0; nt < 4; ++nt) {
      v8f a = zf;
      a = wmma_bf16(aq0, load_frag_b(Kt + 0 * 64 + nt * 16, 64), a);
      a = wmma_bf16(aq1, load_frag_b(Kt + 32 * 64 + nt * 16, 64), a);
#pragma unroll
      for (int r = 0; r < 8; ++r) s[nt][r] = a[r] * scale;
    }

    if (kb == qb) {                     // causal mask only on diagonal block
#pragma unroll
      for (int nt = 0; nt < 4; ++nt) {
#pragma unroll
        for (int r = 0; r < 8; ++r) {
          int qrow = q0 + wave * 16 + r + 8 * hi;
          int kcol = kb * 64 + nt * 16 + nl;
          if (kcol > qrow) s[nt][r] = -1e30f;
        }
      }
    }

    // online softmax: row max -> p -> row sum -> rescale O
    float mnew[8];
#pragma unroll
    for (int r = 0; r < 8; ++r) {
      float v = fmaxf(fmaxf(s[0][r], s[1][r]), fmaxf(s[2][r], s[3][r]));
#pragma unroll
      for (int off = 1; off < 16; off <<= 1) v = fmaxf(v, __shfl_xor(v, off, 32));
      mnew[r] = fmaxf(mrow[r], v);
    }

    unsigned short* Pw = Ps + wave * 16 * 64;
    float psum[8];
#pragma unroll
    for (int r = 0; r < 8; ++r) psum[r] = 0.f;
#pragma unroll
    for (int nt = 0; nt < 4; ++nt) {
#pragma unroll
      for (int r = 0; r < 8; ++r) {
        float p = __expf(s[nt][r] - mnew[r]);
        psum[r] += p;
        Pw[(r + 8 * hi) * 64 + nt * 16 + nl] = f2bf(p);
      }
    }
#pragma unroll
    for (int r = 0; r < 8; ++r) {
      float v = psum[r];
#pragma unroll
      for (int off = 1; off < 16; off <<= 1) v += __shfl_xor(v, off, 32);
      float corr = __expf(mrow[r] - mnew[r]);
      lrow[r] = lrow[r] * corr + v;
      mrow[r] = mnew[r];
      o[0][r] *= corr; o[1][r] *= corr; o[2][r] *= corr; o[3][r] *= corr;
    }

    // O += P V  (per-wave LDS round-trip converts C-layout -> A-layout)
    v16bf ap0 = load_frag_a(Pw + 0, 64);
    v16bf ap1 = load_frag_a(Pw + 32, 64);
#pragma unroll
    for (int nt = 0; nt < 4; ++nt) {
      o[nt] = wmma_bf16(ap0, load_frag_b(Vs + 0 * 64 + nt * 16, 64), o[nt]);
      o[nt] = wmma_bf16(ap1, load_frag_b(Vs + 32 * 64 + nt * 16, 64), o[nt]);
    }
    __syncthreads();
  }

  // epilogue: normalize and write to y[b][t][h*64+d] (bf16)
#pragma unroll
  for (int nt = 0; nt < 4; ++nt) {
#pragma unroll
    for (int r = 0; r < 8; ++r) {
      int t = q0 + wave * 16 + r + 8 * hi;
      int d = nt * 16 + nl;
      float val = o[nt][r] / lrow[r];
      Y[((size_t)b * TSEQ + t) * CEMB + h * HD + d] = f2bf(val);
    }
  }
}

// ---------------- Output projection: [8192,768] x [768,768] -> fp32 out -------
__global__ void __launch_bounds__(128)
gemm_proj(const unsigned short* __restrict__ X, const unsigned short* __restrict__ W,
          float* __restrict__ Out) {
  __shared__ __attribute__((aligned(16))) unsigned short As[128 * 32];
  __shared__ __attribute__((aligned(16))) unsigned short Bs[32 * 64];
  const int tid = threadIdx.x, lane = tid & 31, wave = tid >> 5;
  const int m0 = blockIdx.y * 128, n0 = blockIdx.x * 64;

  const v8f zf = {0.f, 0.f, 0.f, 0.f, 0.f, 0.f, 0.f, 0.f};
  v8f acc[2][4];
#pragma unroll
  for (int mt = 0; mt < 2; ++mt)
#pragma unroll
    for (int nt = 0; nt < 4; ++nt) acc[mt][nt] = zf;

  for (int kt = 0; kt < CEMB / 32; ++kt) {
#pragma unroll
    for (int i = 0; i < 4; ++i) {
      int q = tid + i * 128;
      int r = q >> 2, c = q & 3;
      async_ld_b128(As + r * 32 + c * 8, X + (size_t)(m0 + r) * CEMB + kt * 32 + c * 8);
    }
#pragma unroll
    for (int i = 0; i < 2; ++i) {
      int q = tid + i * 128;
      int r = q >> 3, c = q & 7;
      async_ld_b128(Bs + r * 64 + c * 8, W + (size_t)(kt * 32 + r) * CEMB + n0 + c * 8);
    }
    if (kt + 1 < CEMB / 32)
      __builtin_prefetch((const void*)(W + (size_t)((kt + 1) * 32) * CEMB + n0 + lane * 8), 0, 1);
    wait_async0();
    __syncthreads();

    v16bf a0 = load_frag_a(As + (wave * 32 +  0) * 32, 32);
    v16bf a1 = load_frag_a(As + (wave * 32 + 16) * 32, 32);
#pragma unroll
    for (int nt = 0; nt < 4; ++nt) {
      v16bf b = load_frag_b(Bs + nt * 16, 64);
      acc[0][nt] = wmma_bf16(a0, b, acc[0][nt]);
      acc[1][nt] = wmma_bf16(a1, b, acc[1][nt]);
    }
    __syncthreads();
  }

  const int hi = lane >> 4, nl = lane & 15;
#pragma unroll
  for (int mt = 0; mt < 2; ++mt)
#pragma unroll
    for (int nt = 0; nt < 4; ++nt)
#pragma unroll
      for (int r = 0; r < 8; ++r) {
        int m = m0 + wave * 32 + mt * 16 + r + 8 * hi;
        int n = n0 + nt * 16 + nl;
        Out[(size_t)m * CEMB + n] = acc[mt][nt][r];
      }
}

// ---------------- launch ----------------
extern "C" void kernel_launch(void* const* d_in, const int* in_sizes, int n_in,
                              void* d_out, int out_size, void* d_ws, size_t ws_size,
                              hipStream_t stream) {
  const float* x      = (const float*)d_in[0];   // [8,1024,768]
  const float* w_attn = (const float*)d_in[1];   // [768,2304]
  const float* w_proj = (const float*)d_in[2];   // [768,768]
  float* out = (float*)d_out;                    // [8,1024,768]

  unsigned short* ws = (unsigned short*)d_ws;
  size_t off = 0;
  auto carve = [&](size_t n) {
    unsigned short* p = ws + off;
    off += (n + 127) & ~(size_t)127;
    return p;
  };
  unsigned short* wab = carve((size_t)CEMB * C3);          // bf16 w_attn
  unsigned short* wpb = carve((size_t)CEMB * CEMB);        // bf16 w_proj
  unsigned short* xb  = carve((size_t)MROWS * CEMB);       // bf16 x
  unsigned short* qb  = carve((size_t)BB * NH * TSEQ * HD);
  unsigned short* kb  = carve((size_t)BB * NH * TSEQ * HD);
  unsigned short* vb  = carve((size_t)BB * NH * TSEQ * HD);
  unsigned short* yb  = carve((size_t)MROWS * CEMB);       // bf16 attn output

  cvt_bf16<<<(MROWS * CEMB) / 256, 256, 0, stream>>>(x, xb);
  cvt_bf16<<<(CEMB * C3) / 256, 256, 0, stream>>>(w_attn, wab);
  cvt_bf16<<<(CEMB * CEMB) / 256, 256, 0, stream>>>(w_proj, wpb);

  gemm_qkv<<<dim3(C3 / 64, MROWS / 128), 128, 0, stream>>>(xb, wab, qb, kb, vb);
  attn_flash<<<dim3(TSEQ / 64, BB * NH), 128, 0, stream>>>(qb, kb, vb, yb);
  gemm_proj<<<dim3(CEMB / 64, MROWS / 128), 128, 0, stream>>>(yb, wpb, out);
}
